// InterpersonalGraph_33981781246186
// MI455X (gfx1250) — matrix operand order repaired and checked by reading, not compiled, not run on hardware
//
#include <hip/hip_runtime.h>

typedef __attribute__((ext_vector_type(16))) _Float16 v16h;
typedef __attribute__((ext_vector_type(8)))  _Float16 v8h;
typedef __attribute__((ext_vector_type(8)))  float    v8f;

#define NN    25
#define DIM   128
#define KNNB  4
#define HID   64
#define EROWS 112            // 100 edge rows padded to 7*16
#define EK    288            // 259 padded to 9*32
#define NK    256

// workspace layout (in halves): transposed f16 weights Wt[ncol][K]
#define WS_EW1T 0
#define WS_NW1T (WS_EW1T + 64 * EK)     // 18432
#define WS_EW2T (WS_NW1T + 64 * NK)     // 34816
#define WS_NW2T (WS_EW2T + 128 * 64)    // 43008
#define WS_TOTAL (WS_NW2T + 128 * 64)   // 51200 halves = 100 KiB

__device__ __forceinline__ v8f zero8f() {
  v8f z;
#pragma unroll
  for (int t = 0; t < 8; ++t) z[t] = 0.f;
  return z;
}
__device__ __forceinline__ v16h cat8(v8h lo, v8h hi) {
  v16h r;
#pragma unroll
  for (int t = 0; t < 8; ++t) { r[t] = lo[t]; r[8 + t] = hi[t]; }
  return r;
}
// A fragment from a row-major f16 row pointer: lane-half h takes K {8h..8h+7, 16+8h..+7}
__device__ __forceinline__ v16h fragA(const _Float16* row, int kbase, int h) {
  v8h lo = *(const v8h*)(row + kbase + 8 * h);
  v8h hi = *(const v8h*)(row + kbase + 16 + 8 * h);
  return cat8(lo, hi);
}
// B fragment: lane (n,h) holds Wt[col][kbase + {8h..8h+7, 16+8h..16+8h+7}]
__device__ __forceinline__ v16h fragB(const _Float16* Wt, int col, int kbase, int stride, int h) {
  const _Float16* p = Wt + col * stride + kbase + 8 * h;
  v8h lo = *(const v8h*)p;
  v8h hi = *(const v8h*)(p + 16);
  return cat8(lo, hi);
}

// ---------------------------------------------------------------------------
// Kernel 0: transpose + f16-quantize weights into workspace (run once/launch)
// ---------------------------------------------------------------------------
__global__ void ig_prep_weights(const float* __restrict__ ew1, const float* __restrict__ nw1,
                                const float* __restrict__ ew2, const float* __restrict__ nw2,
                                _Float16* __restrict__ ws) {
  const int t = blockIdx.x * blockDim.x + threadIdx.x;
  const int stride = gridDim.x * blockDim.x;
  _Float16* ew1t = ws + WS_EW1T;
  _Float16* nw1t = ws + WS_NW1T;
  _Float16* ew2t = ws + WS_EW2T;
  _Float16* nw2t = ws + WS_NW2T;
  for (int idx = t; idx < 64 * EK; idx += stride) {
    int c = idx / EK, k = idx % EK;
    ew1t[idx] = (k < 2 * DIM + 3) ? (_Float16)ew1[k * 64 + c] : (_Float16)0.f;
  }
  for (int idx = t; idx < 64 * NK; idx += stride) {
    int c = idx / NK, k = idx % NK;
    nw1t[idx] = (_Float16)nw1[k * 64 + c];
  }
  for (int idx = t; idx < 128 * 64; idx += stride) {
    int c = idx / 64, k = idx % 64;
    ew2t[idx] = (_Float16)ew2[k * 128 + c];
  }
  for (int idx = t; idx < 128 * 64; idx += stride) {
    int c = idx / 64, k = idx % 64;
    nw2t[idx] = (_Float16)nw2[k * 128 + c];
  }
}

// ---------------------------------------------------------------------------
// Kernel 1: one fused graph step per (b,t). 256 threads = 8 wave32.
// ---------------------------------------------------------------------------
__global__ __launch_bounds__(256) void ig_graph(
    const float* __restrict__ emb, const float* __restrict__ bboxes,
    const unsigned char* __restrict__ pmask,
    const float* __restrict__ eb1, const float* __restrict__ eb2,
    const float* __restrict__ nb1, const float* __restrict__ nb2,
    const float* __restrict__ lng, const float* __restrict__ lnb,
    const _Float16* __restrict__ ws, float* __restrict__ out) {
  __shared__ __align__(16) _Float16 s_xh[32 * DIM];      // x in f16 (rows 25..31 zero)
  __shared__ __align__(16) _Float16 s_aggh[32 * DIM];    // agg in f16
  __shared__ __align__(16) _Float16 s_H[EROWS * HID];    // edge hidden; reused as node hidden
  __shared__ __align__(16) float    s_fbuf[32 * DIM];    // agg f32, then delta f32
  __shared__ __align__(16) _Float16 s_ech[EROWS * 32];   // K-slice 256..287 of edge input (3 real)
  __shared__ float s_box[32 * 4];
  __shared__ float s_mask[32];
  __shared__ int   s_nbrj[EROWS];
  __shared__ float s_validf[EROWS];
  __shared__ float s_invden[32];
  __shared__ float s_hasnbr[32];

  const int bt   = blockIdx.x;
  const int tid  = threadIdx.x;
  const int w    = __builtin_amdgcn_readfirstlane(tid >> 5);  // wave id, force SGPR
  const int lane = tid & 31;
  const int n    = lane & 15;   // column / row-in-tile index
  const int h    = lane >> 4;   // lane half

  const _Float16* ew1t = ws + WS_EW1T;
  const _Float16* nw1t = ws + WS_NW1T;
  const _Float16* ew2t = ws + WS_EW2T;
  const _Float16* nw2t = ws + WS_NW2T;

  // ---- stage 0: zero padded LDS, load boxes/mask ----
  for (int idx = tid; idx < 32 * DIM; idx += 256) {
    s_fbuf[idx] = 0.f;
    s_xh[idx]   = (_Float16)0.f;
  }
  for (int idx = tid; idx < EROWS * 32; idx += 256) s_ech[idx] = (_Float16)0.f;
  if (tid < NN * 4) s_box[tid] = bboxes[bt * NN * 4 + tid];
  if (tid < NN)     s_mask[tid] = pmask[bt * NN + tid] ? 1.f : 0.f;
  __syncthreads();

  // ---- stage 1: x -> f16 LDS, KNN + edge features (one lane per node) ----
  for (int idx = tid; idx < NN * DIM; idx += 256)
    s_xh[idx] = (_Float16)emb[bt * NN * DIM + idx];

  if (tid < NN) {
    const int i = tid;
    const float cx = s_box[i * 4 + 0], cy = s_box[i * 4 + 1];
    const float hscale = fmaxf(s_box[i * 4 + 3], 1e-6f);
    const float mi = s_mask[i];
    float bd[KNNB] = {1e6f, 1e6f, 1e6f, 1e6f};
    int   bj[KNNB] = {0, 0, 0, 0};
    for (int j = 0; j < NN; ++j) {
      if (j == i) continue;
      float dx = cx - s_box[j * 4 + 0];
      float dy = cy - s_box[j * 4 + 1];
      float d = sqrtf(dx * dx + dy * dy + 1e-6f) / hscale;
      if (!(mi > 0.f && s_mask[j] > 0.f)) d = 1e6f;
      if (d < bd[3]) {                      // stable 4-way insertion (ties keep earlier j)
        if (d < bd[0])      { bd[3]=bd[2]; bj[3]=bj[2]; bd[2]=bd[1]; bj[2]=bj[1]; bd[1]=bd[0]; bj[1]=bj[0]; bd[0]=d; bj[0]=j; }
        else if (d < bd[1]) { bd[3]=bd[2]; bj[3]=bj[2]; bd[2]=bd[1]; bj[2]=bj[1]; bd[1]=d; bj[1]=j; }
        else if (d < bd[2]) { bd[3]=bd[2]; bj[3]=bj[2]; bd[2]=d; bj[2]=j; }
        else                { bd[3]=d; bj[3]=j; }
      }
    }
    float cnt = 0.f;
    for (int k = 0; k < KNNB; ++k) {
      const int row = i * KNNB + k;
      const int j = bj[k];
      const float valid = (bd[k] < 2.5f) ? 1.f : 0.f;
      cnt += valid;
      s_validf[row] = valid;
      s_nbrj[row] = j;
      float dxn = (cx - s_box[j * 4 + 0]) / hscale;
      float dyn = (cy - s_box[j * 4 + 1]) / hscale;
      float dn  = bd[k];
      if (valid == 0.f) { dxn = 0.f; dyn = 0.f; dn = 0.f; }  // keep f16 finite
      s_ech[row * 32 + 0] = (_Float16)dxn;
      s_ech[row * 32 + 1] = (_Float16)dyn;
      s_ech[row * 32 + 2] = (_Float16)dn;
    }
    s_invden[i] = 1.f / fmaxf(cnt, 1.f);
    s_hasnbr[i] = (cnt > 0.f) ? 1.f : 0.f;
  }
  if (tid < EROWS - NN * KNNB) {            // 12 padding edge rows: point at zeroed data
    const int row = NN * KNNB + tid;
    s_validf[row] = 0.f;
    s_nbrj[row] = 0;
  }
  if (tid < 32 - NN) { const int i = NN + tid; s_invden[i] = 1.f; s_hasnbr[i] = 0.f; }
  __syncthreads();

  // ---- stage 2: edge MLP layer 1 : [112 x 288] @ [288 x 64] ----
  // Padding rows (100..111) are structurally valid: row>>2 <= 27 hits zeroed x rows,
  // nbrj = 0, feature slice zeroed -> every A-fragment load is unconditional.
  {
    const int ntH = w & 3;
    const int col = 16 * ntH + n;
    v16h Bf[9];                              // register-resident B column tile
#pragma unroll
    for (int s = 0; s < 9; ++s) Bf[s] = fragB(ew1t, col, 32 * s, EK, h);
    const float b1 = eb1[col];
    for (int m = (w >> 2); m < 7; m += 2) {  // scalar loop: waves 0-3 even m, 4-7 odd m
      const int rowl = 16 * m + n;
      const _Float16* xi = s_xh + (rowl >> 2) * DIM;
      const _Float16* xj = s_xh + s_nbrj[rowl] * DIM;
      const _Float16* ep = s_ech + rowl * 32;
      v8f c = zero8f();
#pragma unroll
      for (int s = 0; s < 4; ++s)            // K 0..127   : x_i
        c = __builtin_amdgcn_wmma_f32_16x16x32_f16(false, fragA(xi, 32 * s, h), false, Bf[s],
                                                   (short)0, c, false, false);
#pragma unroll
      for (int s = 0; s < 4; ++s)            // K 128..255 : x_j
        c = __builtin_amdgcn_wmma_f32_16x16x32_f16(false, fragA(xj, 32 * s, h), false, Bf[4 + s],
                                                   (short)0, c, false, false);
      // K 256..287 : geometric features (zero-padded)
      c = __builtin_amdgcn_wmma_f32_16x16x32_f16(false, fragA(ep, 0, h), false, Bf[8],
                                                 (short)0, c, false, false);
#pragma unroll
      for (int r = 0; r < 8; ++r) {
        float v = c[r] + b1;
        v = v > 0.f ? v : 0.f;               // ReLU
        s_H[(16 * m + 8 * h + r) * HID + col] = (_Float16)v;
      }
    }
  }
  __syncthreads();

  // ---- stage 3: edge MLP layer 2 [112x64]@[64x128] + masked mean over 4 nbrs ----
  {
    const int col = 16 * w + n;
    const v16h B0 = fragB(ew2t, col, 0, 64, h);
    const v16h B1 = fragB(ew2t, col, 32, 64, h);
    const float b2 = eb2[col];
    for (int m = 0; m < 7; ++m) {
      const _Float16* hr = s_H + (16 * m + n) * HID;
      v8f c = zero8f();
      c = __builtin_amdgcn_wmma_f32_16x16x32_f16(false, fragA(hr, 0, h), false, B0, (short)0, c, false, false);
      c = __builtin_amdgcn_wmma_f32_16x16x32_f16(false, fragA(hr, 32, h), false, B1, (short)0, c, false, false);
      // lane owns rows rbase..rbase+7 = two complete 4-neighbor groups
      const int rbase = 16 * m + 8 * h;
      float s0 = 0.f, s1 = 0.f;
#pragma unroll
      for (int r = 0; r < 4; ++r) s0 += (c[r] + b2) * s_validf[rbase + r];
#pragma unroll
      for (int r = 4; r < 8; ++r) s1 += (c[r] + b2) * s_validf[rbase + r];
      const int i0 = 4 * m + 2 * h;
      s_fbuf[i0 * DIM + col]       = s0 * s_invden[i0];
      s_fbuf[(i0 + 1) * DIM + col] = s1 * s_invden[i0 + 1];
    }
  }
  __syncthreads();

  // agg f32 -> f16 (rows >= 28 were zero-initialized and never written)
  for (int idx = tid; idx < 32 * DIM; idx += 256) s_aggh[idx] = (_Float16)s_fbuf[idx];
  __syncthreads();

  // ---- stage 4: node MLP layer 1 : [32 x 256] @ [256 x 64] (one tile per wave) ----
  {
    const int m = w & 1;
    const int ntH = w >> 1;
    const int col = 16 * ntH + n;
    const float b1 = nb1[col];
    const int rowl = 16 * m + n;
    const _Float16* xr = s_xh + rowl * DIM;
    const _Float16* ar = s_aggh + rowl * DIM;
    v8f c = zero8f();
#pragma unroll
    for (int s = 0; s < 4; ++s)              // K 0..127   : x
      c = __builtin_amdgcn_wmma_f32_16x16x32_f16(false, fragA(xr, 32 * s, h), false,
                                                 fragB(nw1t, col, 32 * s, NK, h),
                                                 (short)0, c, false, false);
#pragma unroll
    for (int s = 0; s < 4; ++s)              // K 128..255 : agg
      c = __builtin_amdgcn_wmma_f32_16x16x32_f16(false, fragA(ar, 32 * s, h), false,
                                                 fragB(nw1t, col, 128 + 32 * s, NK, h),
                                                 (short)0, c, false, false);
#pragma unroll
    for (int r = 0; r < 8; ++r) {
      float v = c[r] + b1;
      v = v > 0.f ? v : 0.f;
      s_H[(16 * m + 8 * h + r) * HID + col] = (_Float16)v;  // reuse s_H as node hidden
    }
  }
  __syncthreads();

  // ---- stage 5: node MLP layer 2 [32x64]@[64x128] -> delta ----
  {
    const int col = 16 * w + n;
    const v16h B0 = fragB(nw2t, col, 0, 64, h);
    const v16h B1 = fragB(nw2t, col, 32, 64, h);
    const float b2 = nb2[col];
#pragma unroll
    for (int m = 0; m < 2; ++m) {
      const _Float16* hr = s_H + (16 * m + n) * HID;
      v8f c = zero8f();
      c = __builtin_amdgcn_wmma_f32_16x16x32_f16(false, fragA(hr, 0, h), false, B0, (short)0, c, false, false);
      c = __builtin_amdgcn_wmma_f32_16x16x32_f16(false, fragA(hr, 32, h), false, B1, (short)0, c, false, false);
      const int rbase = 16 * m + 8 * h;
#pragma unroll
      for (int r = 0; r < 8; ++r) {
        const int row = rbase + r;
        s_fbuf[row * DIM + col] = (c[r] + b2) * s_hasnbr[row];  // fbuf now holds delta
      }
    }
  }
  __syncthreads();

  // ---- stage 6: residual + LayerNorm + mask (one lane per node) ----
  if (tid < NN) {
    const float* xr = emb + bt * NN * DIM + tid * DIM;
    const float* dr = s_fbuf + tid * DIM;
    float mu = 0.f, m2 = 0.f;
    for (int c = 0; c < DIM; ++c) { float y = xr[c] + dr[c]; mu += y; m2 += y * y; }
    mu *= (1.f / DIM);
    const float var = m2 * (1.f / DIM) - mu * mu;
    const float inv = rsqrtf(var + 1e-5f);
    const float mk = s_mask[tid];
    float* orow = out + bt * NN * DIM + tid * DIM;
    for (int c = 0; c < DIM; ++c) {
      float y = xr[c] + dr[c];
      orow[c] = ((y - mu) * inv * lng[c] + lnb[c]) * mk;
    }
  }
}

extern "C" void kernel_launch(void* const* d_in, const int* in_sizes, int n_in,
                              void* d_out, int out_size, void* d_ws, size_t ws_size,
                              hipStream_t stream) {
  (void)n_in; (void)out_size; (void)ws_size;
  const float* emb            = (const float*)d_in[0];
  const float* bboxes         = (const float*)d_in[1];
  const unsigned char* pmask  = (const unsigned char*)d_in[2];
  const float* ew1            = (const float*)d_in[3];
  const float* eb1            = (const float*)d_in[4];
  const float* ew2            = (const float*)d_in[5];
  const float* eb2            = (const float*)d_in[6];
  const float* nw1            = (const float*)d_in[7];
  const float* nb1            = (const float*)d_in[8];
  const float* nw2            = (const float*)d_in[9];
  const float* nb2            = (const float*)d_in[10];
  const float* lng            = (const float*)d_in[11];
  const float* lnb            = (const float*)d_in[12];

  const int BT = in_sizes[0] / (NN * DIM);   // 2048

  ig_prep_weights<<<64, 256, 0, stream>>>(ew1, nw1, ew2, nw2, (_Float16*)d_ws);
  ig_graph<<<BT, 256, 0, stream>>>(emb, bboxes, pmask, eb1, eb2, nb1, nb2, lng, lnb,
                                   (const _Float16*)d_ws, (float*)d_out);
}
